// SwitchTransformerRouting_15006615733154
// MI455X (gfx1250) — matrix operand order repaired and checked by reading, not compiled, not run on hardware
//
#include <hip/hip_runtime.h>
#include <math.h>

#define DIM    1024
#define HID    4096
#define NEXP   8
#define NTOK   4096
#define TILE_M 32          // 2 WMMA M sub-tiles per workgroup
#define BLK    512         // 16 waves

typedef __attribute__((ext_vector_type(16))) __bf16 v16bf;
typedef __attribute__((ext_vector_type(8)))  float  v8f;

union FragAB {
    v16bf  v;
    __bf16 h[16];
    unsigned u[8];
};
union PackBF { __bf16 h[2]; unsigned u; };

// branch-free GELU (tanh form): error << bf16 quantization of h
__device__ __forceinline__ float gelu_fast(float x) {
    const float u  = 0.7978845608028654f * x * (1.0f + 0.044715f * x * x);
    const float au = fabsf(u);
    const float e  = __expf(-2.0f * au);                    // v_exp_f32, e in (0,1]
    const float t  = (1.0f - e) * __builtin_amdgcn_rcpf(1.0f + e);
    return 0.5f * x * (1.0f + __builtin_copysignf(t, u));
}
__device__ __forceinline__ v8f wmma_bf16(const FragAB& a, const FragAB& b, const v8f& c) {
    return __builtin_amdgcn_wmma_f32_16x16x32_bf16(
        false, a.v, false, b.v, (short)0, c, false, false);
}

// ---------------------------------------------------------------- init
__global__ void moe_init_kernel(int* __restrict__ counts) {
    if (threadIdx.x < NEXP) counts[threadIdx.x] = 0;
}

// ---------------------------------------------------------------- gate
// one wave (32 lanes) per token; 8 waves per block; fp32 exact routing
__global__ __launch_bounds__(256) void moe_gate_kernel(
    const float* __restrict__ x, const float* __restrict__ gw,
    const float* __restrict__ gb, int* __restrict__ expert_id)
{
    const int t    = blockIdx.x * 8 + (threadIdx.x >> 5);
    const int lane = threadIdx.x & 31;
    const float* xr = x + (size_t)t * DIM;

    float acc[NEXP];
    #pragma unroll
    for (int e = 0; e < NEXP; ++e) acc[e] = 0.0f;

    for (int c = lane; c < DIM; c += 32) {
        const float xv = xr[c];
        #pragma unroll
        for (int e = 0; e < NEXP; ++e) acc[e] += xv * gw[c * NEXP + e];
    }
    #pragma unroll
    for (int e = 0; e < NEXP; ++e) {
        #pragma unroll
        for (int off = 16; off > 0; off >>= 1)
            acc[e] += __shfl_xor(acc[e], off, 32);
    }
    if (lane == 0) {
        float lg[NEXP];
        #pragma unroll
        for (int e = 0; e < NEXP; ++e) lg[e] = acc[e] + gb[e];
        int i1 = 0; float v1 = lg[0];
        #pragma unroll
        for (int e = 1; e < NEXP; ++e) if (lg[e] > v1) { v1 = lg[e]; i1 = e; }
        int i2 = -1; float v2 = -3.4e38f;
        #pragma unroll
        for (int e = 0; e < NEXP; ++e) {
            if (e == i1) continue;
            if (lg[e] > v2) { v2 = lg[e]; i2 = e; }
        }
        expert_id[t] = (i1 > i2) ? i1 : i2;   // quirk: max of top-2 indices wins
    }
}

// ---------------------------------------------------------------- bucket
__global__ __launch_bounds__(256) void moe_bucket_kernel(
    const int* __restrict__ expert_id, int* __restrict__ counts,
    int* __restrict__ token_idx)
{
    const int t = blockIdx.x * 256 + threadIdx.x;
    if (t < NTOK) {
        const int e = expert_id[t];
        const int slot = atomicAdd(&counts[e], 1);
        token_idx[e * NTOK + slot] = t;
    }
}

// ---------------------------------------------------------------- fused FFN
// grid = (NTOK/TILE_M, NEXP); block = 512 (16 waves).
// GEMM1: wave wv computes h cols [hc*256 + wv*16, +16) for both M sub-tiles.
// GEMM2: wave wv owns Y cols [wv*64, +64) (4 n-tiles) x 2 M sub-tiles.
__global__ __launch_bounds__(BLK) void moe_ffn_kernel(
    const float* __restrict__ x,
    const float* __restrict__ w1, const float* __restrict__ b1,
    const float* __restrict__ w2, const float* __restrict__ b2,
    const int* __restrict__ counts, const int* __restrict__ token_idx,
    float* __restrict__ out)
{
    // padded dword strides keep ds_load_b128 rows bank-spread, 16B aligned
    __shared__ __align__(16) unsigned x_lds[TILE_M * 516];  // 32 x 1024 bf16
    __shared__ __align__(16) unsigned h_lds[TILE_M * 132];  // 32 x  256 bf16
    __shared__ int tok_s[TILE_M];

    const int e    = blockIdx.y;
    const int cnt  = counts[e];
    const int base = blockIdx.x * TILE_M;
    if (base >= cnt) return;                   // uniform branch: EXEC all-ones below
    const int mcount = min(TILE_M, cnt - base);

    const int tid  = threadIdx.x;
    const int lane = tid & 31;
    const int wv   = tid >> 5;                 // 0..15
    const int ncol = lane & 15;
    const int kb_b  = (lane < 16) ? 0 : 8;     // B-frag K base (table order: 8h, +16 for upper half)
    const int a_off = (lane < 16) ? 0 : 4;     // A-frag dword base (K base 0/8)

    if (tid < TILE_M) {
        const int slot = base + ((tid < mcount) ? tid : 0);  // pad with tile base
        tok_s[tid] = token_idx[e * NTOK + slot];
    }
    __syncthreads();

    // ---- stage X tile as packed bf16 in LDS (coalesced float2 loads)
    for (int idx = tid; idx < TILE_M * (DIM / 2); idx += BLK) {
        const int m  = idx >> 9;        // / 512
        const int dp = idx & 511;
        const float2 xv =
            *reinterpret_cast<const float2*>(&x[(size_t)tok_s[m] * DIM + dp * 2]);
        PackBF p; p.h[0] = (__bf16)xv.x; p.h[1] = (__bf16)xv.y;
        x_lds[m * 516 + dp] = p.u;
    }
    __syncthreads();

    const float* w1e = w1 + (size_t)e * DIM * HID;
    const float* w2e = w2 + (size_t)e * HID * DIM;

    const v8f zero = {0.f, 0.f, 0.f, 0.f, 0.f, 0.f, 0.f, 0.f};
    v8f yacc[2][4];
    #pragma unroll
    for (int s = 0; s < 2; ++s)
        #pragma unroll
        for (int j = 0; j < 4; ++j) yacc[s][j] = zero;

    __bf16* hls = reinterpret_cast<__bf16*>(h_lds);

    for (int hc = 0; hc < HID / 256; ++hc) {
        // ============ GEMM1: h[32, 16] for this wave's n-tile ==============
        v8f hacc[2] = {zero, zero};
        const int n0 = hc * 256 + wv * 16;
        for (int k0 = 0; k0 < DIM; k0 += 32) {
            FragAB a0, a1, b;
            const unsigned* r0 = &x_lds[ncol * 516 + (k0 >> 1) + a_off];
            const unsigned* r1 = r0 + 16 * 516;
            *reinterpret_cast<uint4*>(&a0.u[0]) = *reinterpret_cast<const uint4*>(&r0[0]);
            *reinterpret_cast<uint4*>(&a0.u[4]) = *reinterpret_cast<const uint4*>(&r0[8]);
            *reinterpret_cast<uint4*>(&a1.u[0]) = *reinterpret_cast<const uint4*>(&r1[0]);
            *reinterpret_cast<uint4*>(&a1.u[4]) = *reinterpret_cast<const uint4*>(&r1[8]);
            #pragma unroll
            for (int i = 0; i < 8; ++i) {
                const size_t k = (size_t)(k0 + kb_b + 2 * i + ((i >> 2) << 3));
                b.h[2 * i]     = (__bf16)w1e[k * HID + n0 + ncol];
                b.h[2 * i + 1] = (__bf16)w1e[(k + 1) * HID + n0 + ncol];
            }
            hacc[0] = wmma_bf16(a0, b, hacc[0]);
            hacc[1] = wmma_bf16(a1, b, hacc[1]);
        }

        // bias + fast GELU, round-trip h chunk through LDS as bf16
        const float bias = b1[(size_t)e * HID + n0 + ncol];
        const int mbase = (lane < 16) ? 0 : 8;
        #pragma unroll
        for (int s = 0; s < 2; ++s)
            #pragma unroll
            for (int v = 0; v < 8; ++v) {
                const int m = s * 16 + mbase + v;
                hls[m * 264 + (wv * 16 + ncol)] = (__bf16)gelu_fast(hacc[s][v] + bias);
            }
        __syncthreads();

        // ============ GEMM2: Y += h_chunk @ w2 =============================
        for (int kk = 0; kk < 256; kk += 32) {
            FragAB a0, a1;
            const unsigned* r0 = &h_lds[ncol * 132 + (kk >> 1) + a_off];
            const unsigned* r1 = r0 + 16 * 132;
            *reinterpret_cast<uint4*>(&a0.u[0]) = *reinterpret_cast<const uint4*>(&r0[0]);
            *reinterpret_cast<uint4*>(&a0.u[4]) = *reinterpret_cast<const uint4*>(&r0[8]);
            *reinterpret_cast<uint4*>(&a1.u[0]) = *reinterpret_cast<const uint4*>(&r1[0]);
            *reinterpret_cast<uint4*>(&a1.u[4]) = *reinterpret_cast<const uint4*>(&r1[8]);

            const size_t kg = (size_t)(hc * 256 + kk + kb_b);
            #pragma unroll
            for (int j = 0; j < 4; ++j) {
                const int n = wv * 64 + j * 16 + ncol;
                FragAB b;
                #pragma unroll
                for (int i = 0; i < 8; ++i) {
                    const size_t k = kg + 2 * i + ((i >> 2) << 3);
                    b.h[2 * i]     = (__bf16)w2e[k * DIM + n];
                    b.h[2 * i + 1] = (__bf16)w2e[(k + 1) * DIM + n];
                }
                yacc[0][j] = wmma_bf16(a0, b, yacc[0][j]);
                yacc[1][j] = wmma_bf16(a1, b, yacc[1][j]);
            }
        }
        __syncthreads();   // before next chunk overwrites h_lds
    }

    // ---- epilogue: out[t] = 2 * (Y + b2)   (TOP_K = 2)
    const int mb = (lane < 16) ? 0 : 8;
    #pragma unroll
    for (int s = 0; s < 2; ++s)
        #pragma unroll
        for (int j = 0; j < 4; ++j) {
            const int n = wv * 64 + j * 16 + ncol;
            const float bias = b2[(size_t)e * DIM + n];
            #pragma unroll
            for (int v = 0; v < 8; ++v) {
                const int m = s * 16 + mb + v;
                if (m < mcount)
                    out[(size_t)tok_s[m] * DIM + n] = 2.0f * (yacc[s][j][v] + bias);
            }
        }
}

// ---------------------------------------------------------------- launch
extern "C" void kernel_launch(void* const* d_in, const int* in_sizes, int n_in,
                              void* d_out, int out_size, void* d_ws, size_t ws_size,
                              hipStream_t stream) {
    const float* x      = (const float*)d_in[0];
    const float* gate_w = (const float*)d_in[1];
    const float* gate_b = (const float*)d_in[2];
    const float* w1     = (const float*)d_in[3];
    const float* b1     = (const float*)d_in[4];
    const float* w2     = (const float*)d_in[5];
    const float* b2     = (const float*)d_in[6];
    float* out = (float*)d_out;

    int* counts    = (int*)d_ws;
    int* expert_id = (int*)((char*)d_ws + 64);
    int* token_idx = (int*)((char*)d_ws + 64 + NTOK * sizeof(int));

    moe_init_kernel<<<1, 32, 0, stream>>>(counts);
    moe_gate_kernel<<<NTOK / 8, 256, 0, stream>>>(x, gate_w, gate_b, expert_id);
    moe_bucket_kernel<<<NTOK / 256, 256, 0, stream>>>(expert_id, counts, token_idx);
    moe_ffn_kernel<<<dim3(NTOK / TILE_M, NEXP), dim3(BLK), 0, stream>>>(
        x, w1, b1, w2, b2, counts, token_idx, out);
}
